// GlobalFeatureDualAttention_82910048682143
// MI455X (gfx1250) — compile-verified, matrix operand order checked by reading, not run on hardware
//
#include <hip/hip_runtime.h>

#define B_SZ 16384
#define C_SZ 1024
#define C2_SZ 2048
#define C3_SZ 3072
#define EPSV 1e-5f

typedef __attribute__((ext_vector_type(16))) __bf16 v16bf;
typedef __attribute__((ext_vector_type(8)))  float  v8f;
typedef __attribute__((ext_vector_type(4)))  int    v4i;
typedef unsigned short u16;
typedef unsigned int   u32;

// ---- CDNA5 async memory->LDS path (guarded; falls back to sync staging) ----
#if defined(__has_builtin)
# if __has_builtin(__builtin_amdgcn_global_load_async_to_lds_b128) && \
     __has_builtin(__builtin_amdgcn_s_wait_asynccnt)
#  define USE_ASYNC_LDS 1
# endif
#endif
#ifndef USE_ASYNC_LDS
# define USE_ASYNC_LDS 0
#endif

// builtin signature (from clang diagnostic): (int4 AS1*, int4 AS3*, imm offset, imm cpol)
typedef __attribute__((address_space(1))) v4i* gp128_t;
typedef __attribute__((address_space(3))) v4i* lp128_t;

__device__ __forceinline__ u16 f2bf(float f) {
    union { float f; u32 u; } v; v.f = f;
    u32 r = v.u + 0x7FFFu + ((v.u >> 16) & 1u);   // round-to-nearest-even
    return (u16)(r >> 16);
}

// ---------------- weight f32 -> bf16 conversion (vectorized x4) ----------------
__global__ __launch_bounds__(256) void cvt_bf16_kernel(const float* __restrict__ src,
                                                       u16* __restrict__ dst, int n4) {
    int i = blockIdx.x * 256 + threadIdx.x;
    if (i < n4) {
        float4 f = ((const float4*)src)[i];
        u32 p0 = (u32)f2bf(f.x) | ((u32)f2bf(f.y) << 16);
        u32 p1 = (u32)f2bf(f.z) | ((u32)f2bf(f.w) << 16);
        ((uint2*)dst)[i] = make_uint2(p0, p1);
    }
}

// ---------------- block reduction helper (wave32-aware) ----------------
__device__ __forceinline__ float blockReduceSum(float v, float* sh) {
    #pragma unroll
    for (int o = 16; o > 0; o >>= 1) v += __shfl_down(v, o, 32);
    int lane = threadIdx.x & 31, wv = threadIdx.x >> 5;
    if (lane == 0) sh[wv] = v;
    __syncthreads();
    v = (threadIdx.x < 8) ? sh[threadIdx.x] : 0.0f;
    if (wv == 0) {
        #pragma unroll
        for (int o = 16; o > 0; o >>= 1) v += __shfl_down(v, o, 32);
        if (lane == 0) sh[0] = v;
    }
    __syncthreads();
    float r = sh[0];
    __syncthreads();
    return r;
}

// ---------------- pre: LN(featA row) -> fused[:,0:C], featB -> bf16 ----------------
__global__ __launch_bounds__(256) void pre_ln_kernel(const float* __restrict__ featA,
                                                     const float* __restrict__ featB,
                                                     const float* __restrict__ lw,
                                                     const float* __restrict__ lb,
                                                     u16* __restrict__ fused,
                                                     u16* __restrict__ featB_bf) {
    __shared__ float sh[8];
    int row = blockIdx.x;
    int t = threadIdx.x;
    const float* xa = featA + (size_t)row * C_SZ;
    const float* xb = featB + (size_t)row * C_SZ;

    float4 a4 = *(const float4*)(xa + t * 4);
    float4 b4 = *(const float4*)(xb + t * 4);

    {   // featB -> bf16
        u32 p0 = (u32)f2bf(b4.x) | ((u32)f2bf(b4.y) << 16);
        u32 p1 = (u32)f2bf(b4.z) | ((u32)f2bf(b4.w) << 16);
        *(uint2*)(featB_bf + (size_t)row * C_SZ + t * 4) = make_uint2(p0, p1);
    }

    float s = a4.x + a4.y + a4.z + a4.w;
    float mean = blockReduceSum(s, sh) * (1.0f / (float)C_SZ);
    float d0 = a4.x - mean, d1 = a4.y - mean, d2 = a4.z - mean, d3 = a4.w - mean;
    float var = blockReduceSum(d0*d0 + d1*d1 + d2*d2 + d3*d3, sh) * (1.0f / (float)C_SZ);
    float inv = rsqrtf(var + EPSV);

    float4 w4 = *(const float4*)(lw + t * 4);
    float4 g4 = *(const float4*)(lb + t * 4);
    float o0 = d0 * inv * w4.x + g4.x;
    float o1 = d1 * inv * w4.y + g4.y;
    float o2 = d2 * inv * w4.z + g4.z;
    float o3 = d3 * inv * w4.w + g4.w;
    u32 p0 = (u32)f2bf(o0) | ((u32)f2bf(o1) << 16);
    u32 p1 = (u32)f2bf(o2) | ((u32)f2bf(o3) << 16);
    *(uint2*)(fused + (size_t)row * C3_SZ + t * 4) = make_uint2(p0, p1);
}

// ---------------- mid: LN(V row) with two param sets -> fused[:,C:2C] & [:,2C:3C] ----------------
__global__ __launch_bounds__(256) void mid_ln_kernel(const float* __restrict__ V,
                                                     const float* __restrict__ sw,
                                                     const float* __restrict__ sb,
                                                     const float* __restrict__ dw,
                                                     const float* __restrict__ db,
                                                     u16* __restrict__ fused) {
    __shared__ float sh[8];
    int row = blockIdx.x;
    int t = threadIdx.x;
    const float* xv = V + (size_t)row * C_SZ;
    float4 a4 = *(const float4*)(xv + t * 4);

    float s = a4.x + a4.y + a4.z + a4.w;
    float mean = blockReduceSum(s, sh) * (1.0f / (float)C_SZ);
    float d0 = a4.x - mean, d1 = a4.y - mean, d2 = a4.z - mean, d3 = a4.w - mean;
    float var = blockReduceSum(d0*d0 + d1*d1 + d2*d2 + d3*d3, sh) * (1.0f / (float)C_SZ);
    float inv = rsqrtf(var + EPSV);
    float n0 = d0 * inv, n1 = d1 * inv, n2 = d2 * inv, n3 = d3 * inv;

    float4 w4 = *(const float4*)(sw + t * 4);
    float4 g4 = *(const float4*)(sb + t * 4);
    u32 p0 = (u32)f2bf(n0 * w4.x + g4.x) | ((u32)f2bf(n1 * w4.y + g4.y) << 16);
    u32 p1 = (u32)f2bf(n2 * w4.z + g4.z) | ((u32)f2bf(n3 * w4.w + g4.w) << 16);
    *(uint2*)(fused + (size_t)row * C3_SZ + C_SZ + t * 4) = make_uint2(p0, p1);

    w4 = *(const float4*)(dw + t * 4);
    g4 = *(const float4*)(db + t * 4);
    p0 = (u32)f2bf(n0 * w4.x + g4.x) | ((u32)f2bf(n1 * w4.y + g4.y) << 16);
    p1 = (u32)f2bf(n2 * w4.z + g4.z) | ((u32)f2bf(n3 * w4.w + g4.w) << 16);
    *(uint2*)(fused + (size_t)row * C3_SZ + 2 * C_SZ + t * 4) = make_uint2(p0, p1);
}

// ---------------- WMMA GEMM:  D[M,N] = A[M,K](bf16) x W[N,K]^T(bf16) + epilogue ----------------
// Block tile 128x128, K step 32, 8 waves; each wave owns 32x64 (2x4 WMMA frags = 8 WMMA/step).
// Double-buffered LDS fed by GLOBAL_LOAD_ASYNC_TO_LDS_B128 (ASYNCcnt) when available.
#define TM 128
#define TN 128
#define TK 32
#define LDA 48   // padded LDS row stride (u16) -> 96B rows, 16B aligned, bank-conflict skew

__device__ __forceinline__ void cp16(const u16* g, u16* l) {
#if USE_ASYNC_LDS
    __builtin_amdgcn_global_load_async_to_lds_b128(
        (gp128_t)(uintptr_t)g, (lp128_t)(u32)(uintptr_t)l, 0, 0);
#else
    *(uint4*)l = *(const uint4*)g;
#endif
}

// stage one 128x32 bf16 tile (512 16B chunks, 2 per thread)
__device__ __forceinline__ void stage_tile(const u16* __restrict__ G, int rowBase, int K,
                                           int kt, u16* lds, int tid) {
    #pragma unroll
    for (int j = 0; j < 2; ++j) {
        int c  = tid + j * 256;
        int r  = c >> 2;
        int kc = (c & 3) << 3;
        cp16(&G[(size_t)(rowBase + r) * K + (size_t)kt * TK + kc], &lds[r * LDA + kc]);
    }
}

// Fragment load matching 16-bit A/B VGPR layout:
// lanes 0-15 hold row r, K = {0..7, 16..23}; lanes 16-31 hold row r, K = {8..15, 24..31}
__device__ __forceinline__ v16bf ldfrag(const u16* base, int lane) {
    int r  = lane & 15;
    int kh = (lane >> 4) << 3;
    const u16* p = base + r * LDA + kh;
    union { uint4 q[2]; v16bf v; } u;
    u.q[0] = *(const uint4*)(p);
    u.q[1] = *(const uint4*)(p + 16);
    return u.v;
}

__global__ __launch_bounds__(256) void gemm_wmma_kernel(const u16* __restrict__ A,
                                                        const u16* __restrict__ W,
                                                        int M, int N, int K, int mode,
                                                        float* __restrict__ outF,
                                                        u16* __restrict__ outBf,
                                                        const float* __restrict__ e0,
                                                        const float* __restrict__ e1,
                                                        const float* __restrict__ e2,
                                                        const float* __restrict__ e3) {
    __shared__ u16 lsA[2][TM * LDA];   // 2 x 12 KB
    __shared__ u16 lsB[2][TN * LDA];   // 2 x 12 KB

    int tid  = threadIdx.x;
    int lane = tid & 31;
    int wave = tid >> 5;
    int wm   = wave >> 1;           // 0..3  (M direction, 32 rows each)
    int wn   = wave & 1;            // 0..1  (N direction, 64 cols each)

    int mBase = blockIdx.y * TM;
    int nBase = blockIdx.x * TN;

    v8f acc[2][4] = {};

    int nkt = K / TK;
    // prologue: stage tile 0 into buffer 0
    stage_tile(A, mBase, K, 0, lsA[0], tid);
    stage_tile(W, nBase, K, 0, lsB[0], tid);

    for (int kt = 0; kt < nkt; ++kt) {
        int cur = kt & 1;
        if (kt + 1 < nkt) {
            // issue next tile's copies into the other buffer (4 per thread)
            stage_tile(A, mBase, K, kt + 1, lsA[cur ^ 1], tid);
            stage_tile(W, nBase, K, kt + 1, lsB[cur ^ 1], tid);
#if USE_ASYNC_LDS
            __builtin_amdgcn_s_wait_asynccnt(4);   // current tile's 4 copies complete
#endif
        } else {
#if USE_ASYNC_LDS
            __builtin_amdgcn_s_wait_asynccnt(0);
#endif
        }
        __syncthreads();

        const u16* aB = &lsA[cur][(wm * 32) * LDA];
        const u16* bB = &lsB[cur][(wn * 64) * LDA];
        v16bf af[2], bq[4];
        af[0] = ldfrag(aB, lane);
        af[1] = ldfrag(aB + 16 * LDA, lane);
        #pragma unroll
        for (int j = 0; j < 4; ++j) bq[j] = ldfrag(bB + j * 16 * LDA, lane);

        #pragma unroll
        for (int i = 0; i < 2; ++i)
            #pragma unroll
            for (int j = 0; j < 4; ++j)
                acc[i][j] = __builtin_amdgcn_wmma_f32_16x16x32_bf16(
                    false, af[i], false, bq[j], (short)0, acc[i][j], false, false);

        __syncthreads();
    }

    // Epilogue. D frag layout: lanes 0-15 -> N=lane, M=vgpr; lanes 16-31 -> N=lane-16, M=vgpr+8
    #pragma unroll
    for (int i = 0; i < 2; ++i) {
        #pragma unroll
        for (int j = 0; j < 4; ++j) {
            int col  = nBase + wn * 64 + j * 16 + (lane & 15);
            int row0 = mBase + wm * 32 + i * 16 + ((lane >> 4) << 3);
            v8f a = acc[i][j];
            if (mode == 0) {                          // raw f32 output (V)
                #pragma unroll
                for (int r = 0; r < 8; ++r)
                    outF[(size_t)(row0 + r) * N + col] = a[r];
            } else if (mode == 1) {                   // BN + ReLU -> bf16 (H)
                float sc = e0[col] * rsqrtf(e3[col] + EPSV);   // gamma / sqrt(var+eps)
                float sh = e1[col] - e2[col] * sc;             // beta - mean*scale
                #pragma unroll
                for (int r = 0; r < 8; ++r) {
                    float v = a[r] * sc + sh;
                    outBf[(size_t)(row0 + r) * N + col] = f2bf(v > 0.0f ? v : 0.0f);
                }
            } else {                                  // + b2 + featA residual -> f32 out
                float bb = e0[col];
                #pragma unroll
                for (int r = 0; r < 8; ++r)
                    outF[(size_t)(row0 + r) * N + col] =
                        a[r] + bb + e1[(size_t)(row0 + r) * N + col];
            }
        }
    }
}

extern "C" void kernel_launch(void* const* d_in, const int* in_sizes, int n_in,
                              void* d_out, int out_size, void* d_ws, size_t ws_size,
                              hipStream_t stream) {
    const float* featA    = (const float*)d_in[0];
    const float* featB    = (const float*)d_in[1];
    const float* Wv       = (const float*)d_in[4];
    const float* ln_in_w  = (const float*)d_in[7];
    const float* ln_in_b  = (const float*)d_in[8];
    const float* ln_sim_w = (const float*)d_in[9];
    const float* ln_sim_b = (const float*)d_in[10];
    const float* ln_dis_w = (const float*)d_in[11];
    const float* ln_dis_b = (const float*)d_in[12];
    const float* W1       = (const float*)d_in[13];
    const float* bn_gamma = (const float*)d_in[14];
    const float* bn_beta  = (const float*)d_in[15];
    const float* bn_mean  = (const float*)d_in[16];
    const float* bn_var   = (const float*)d_in[17];
    const float* W2       = (const float*)d_in[18];
    const float* b2       = (const float*)d_in[19];
    float* out = (float*)d_out;

    // workspace carve-up
    char* ws = (char*)d_ws;
    size_t off = 0;
    u16*   featB_bf = (u16*)(ws + off); off += (size_t)B_SZ * C_SZ  * 2;
    u16*   fused    = (u16*)(ws + off); off += (size_t)B_SZ * C3_SZ * 2;
    float* V        = (float*)(ws + off); off += (size_t)B_SZ * C_SZ * 4;
    u16*   Hact     = (u16*)(ws + off); off += (size_t)B_SZ * C2_SZ * 2;
    u16*   Wv_bf    = (u16*)(ws + off); off += (size_t)C_SZ  * C_SZ  * 2;
    u16*   W1_bf    = (u16*)(ws + off); off += (size_t)C2_SZ * C3_SZ * 2;
    u16*   W2_bf    = (u16*)(ws + off); off += (size_t)C_SZ  * C2_SZ * 2;

    // weights -> bf16
    {
        int n4 = (C_SZ * C_SZ) / 4;
        cvt_bf16_kernel<<<(n4 + 255) / 256, 256, 0, stream>>>(Wv, Wv_bf, n4);
        n4 = (C2_SZ * C3_SZ) / 4;
        cvt_bf16_kernel<<<(n4 + 255) / 256, 256, 0, stream>>>(W1, W1_bf, n4);
        n4 = (C_SZ * C2_SZ) / 4;
        cvt_bf16_kernel<<<(n4 + 255) / 256, 256, 0, stream>>>(W2, W2_bf, n4);
    }

    // LN(featA) -> fused[:,0:C]; featB -> bf16
    pre_ln_kernel<<<B_SZ, 256, 0, stream>>>(featA, featB, ln_in_w, ln_in_b, fused, featB_bf);

    // GEMM1: V = featB_bf x Wv^T  (softmax over size-1 axis == 1, so only V survives)
    gemm_wmma_kernel<<<dim3(C_SZ / TN, B_SZ / TM), 256, 0, stream>>>(
        featB_bf, Wv_bf, B_SZ, C_SZ, C_SZ, 0, V, nullptr,
        nullptr, nullptr, nullptr, nullptr);

    // LN(V) twice -> fused[:,C:2C], fused[:,2C:3C]
    mid_ln_kernel<<<B_SZ, 256, 0, stream>>>(V, ln_sim_w, ln_sim_b, ln_dis_w, ln_dis_b, fused);

    // GEMM2: H = relu(BN(fused x W1^T)) -> bf16
    gemm_wmma_kernel<<<dim3(C2_SZ / TN, B_SZ / TM), 256, 0, stream>>>(
        fused, W1_bf, B_SZ, C2_SZ, C3_SZ, 1, nullptr, Hact,
        bn_gamma, bn_beta, bn_mean, bn_var);

    // GEMM3: out = H x W2^T + b2 + featA
    gemm_wmma_kernel<<<dim3(C_SZ / TN, B_SZ / TM), 256, 0, stream>>>(
        Hact, W2_bf, B_SZ, C_SZ, C2_SZ, 2, out, nullptr,
        b2, featA, nullptr, nullptr);
}
